// _AttentionBlock_80169859547963
// MI455X (gfx1250) — matrix operand order repaired and checked by reading, not compiled, not run on hardware
//
#include <hip/hip_runtime.h>
#include <math.h>

// ---------------------------------------------------------------------------
// Transformer encoder block (pre-norm) for MI455X / gfx1250, bf16 WMMA path.
// All matmuls use V_WMMA_F32_16X16X32_BF16 (wave32, 16x16 C/D tiles).
// Global->LDS staging uses the CDNA5 async-to-LDS engine (ASYNCcnt) with
// double-buffered tiles.
// ---------------------------------------------------------------------------

typedef __bf16 bf16_t;
typedef __attribute__((ext_vector_type(16))) __bf16 v16bf;
typedef __attribute__((ext_vector_type(4)))  __bf16 v4bf;
typedef __attribute__((ext_vector_type(8)))  float  v8f;
typedef __attribute__((ext_vector_type(4)))  unsigned int u32x4;
typedef __attribute__((ext_vector_type(4)))  int i32x4;

union Frag16 { v16bf v; u32x4 u[2]; };

__device__ __forceinline__ v8f wmma_bf16(const Frag16& a, const Frag16& b, v8f c) {
  return __builtin_amdgcn_wmma_f32_16x16x32_bf16(false, a.v, false, b.v,
                                                 (short)0, c, false, false);
}

#if __has_builtin(__builtin_amdgcn_global_load_async_to_lds_b128) && \
    __has_builtin(__builtin_amdgcn_s_wait_asynccnt)
#define HAVE_ASYNC_LDS 1
typedef __attribute__((address_space(1))) i32x4 gas_i32x4;
typedef __attribute__((address_space(3))) i32x4 las_i32x4;
#else
#define HAVE_ASYNC_LDS 0
#endif

// copy 16B global -> LDS. Async path: no VGPR staging, tracked by ASYNCcnt.
__device__ __forceinline__ void async_copy16(const bf16_t* g, bf16_t* l) {
#if HAVE_ASYNC_LDS
  __builtin_amdgcn_global_load_async_to_lds_b128(
      (gas_i32x4*)g, (las_i32x4*)l, 0, 0);
#else
  *(u32x4*)l = *(const u32x4*)g;
#endif
}

template <int N>
__device__ __forceinline__ void async_wait() {
#if HAVE_ASYNC_LDS
  __builtin_amdgcn_s_wait_asynccnt(N);
#endif
}

#define D_MODEL 1024
#define N_HEADS 16
#define HEAD_DIM 64
#define FF_DIM 4096
#define SEQ 2048
#define BATCH 2
#define MTOK (BATCH * SEQ)   // 4096 tokens

// ---------------------------------------------------------------------------
// Weight transpose + cast: W[K][N] f32 -> Wt[N][K] bf16  (K-contiguous rows,
// the per-lane-contiguous layout WMMA B-fragments want).
// ---------------------------------------------------------------------------
__global__ __launch_bounds__(256)
void transpose_cast_kernel(const float* __restrict__ W, bf16_t* __restrict__ Wt,
                           int K, int N) {
  __shared__ float tile[32][33];
  const int nb = blockIdx.x * 32, kb = blockIdx.y * 32;
  const int tx = threadIdx.x, ty = threadIdx.y;      // 32 x 8
  #pragma unroll
  for (int i = 0; i < 32; i += 8)
    tile[ty + i][tx] = W[(size_t)(kb + ty + i) * N + nb + tx];
  __syncthreads();
  #pragma unroll
  for (int i = 0; i < 32; i += 8)
    Wt[(size_t)(nb + ty + i) * K + kb + tx] = (bf16_t)tile[tx][ty + i];
}

// ---------------------------------------------------------------------------
// LayerNorm over D_MODEL=1024 + cast to bf16. One row per block, 256 threads.
// ---------------------------------------------------------------------------
__global__ __launch_bounds__(256)
void ln_cast_kernel(const float* __restrict__ x, const float* __restrict__ g,
                    const float* __restrict__ b, bf16_t* __restrict__ out) {
  __shared__ float red[16];
  const int row = blockIdx.x, tid = threadIdx.x;
  const float4 v = ((const float4*)(x + (size_t)row * D_MODEL))[tid];
  float s  = v.x + v.y + v.z + v.w;
  float s2 = v.x * v.x + v.y * v.y + v.z * v.z + v.w * v.w;
  #pragma unroll
  for (int o = 16; o > 0; o >>= 1) {
    s  += __shfl_xor(s,  o, 32);
    s2 += __shfl_xor(s2, o, 32);
  }
  if ((tid & 31) == 0) { red[tid >> 5] = s; red[8 + (tid >> 5)] = s2; }
  __syncthreads();
  s = red[0]; s2 = red[8];
  #pragma unroll
  for (int i = 1; i < 8; i++) { s += red[i]; s2 += red[8 + i]; }
  const float mu   = s * (1.0f / D_MODEL);
  const float var  = s2 * (1.0f / D_MODEL) - mu * mu;
  const float rstd = rsqrtf(var + 1e-5f);
  const float4 gg = ((const float4*)g)[tid];
  const float4 bb = ((const float4*)b)[tid];
  v4bf o;
  o[0] = (bf16_t)((v.x - mu) * rstd * gg.x + bb.x);
  o[1] = (bf16_t)((v.y - mu) * rstd * gg.y + bb.y);
  o[2] = (bf16_t)((v.z - mu) * rstd * gg.z + bb.z);
  o[3] = (bf16_t)((v.w - mu) * rstd * gg.w + bb.w);
  *(v4bf*)(out + (size_t)row * D_MODEL + tid * 4) = o;
}

// ---------------------------------------------------------------------------
// Tiled bf16 GEMM:  C[M,N] = A[M,K] * Bt[N,K]^T + bias, fused epilogues.
// Block tile 128x128, staged K = 64 (two WMMA K-steps per barrier pair ->
// 16 WMMAs between syncs). 256 threads = 8 waves, each wave owns a 64x32
// sub-tile = 4x2 WMMA f32 accumulators. LDS rows padded to 72 bf16 (144B,
// 36-dword stride) so 16B fragment ds_loads are bank-conflict free.
// Double-buffered async-to-LDS staging overlaps stage k+1 with stage k WMMAs
// (8 outstanding b128 async copies per thread, throttled by s_wait_asynccnt).
//   EPI 0: scatter qkv -> Q[B,H,T,64] (pre-scaled 1/8), K, Vt[B,H,64,T] (bf16)
//   EPI 1: outF = resid + C + bias                                     (f32)
//   EPI 2: outB = gelu_exact(C + bias)                                 (bf16)
// ---------------------------------------------------------------------------
template <int EPI>
__global__ __launch_bounds__(256)
void gemm_kernel(const bf16_t* __restrict__ A, const bf16_t* __restrict__ Bt,
                 const float* __restrict__ bias, const float* __restrict__ resid,
                 bf16_t* __restrict__ outQ, bf16_t* __restrict__ outK,
                 bf16_t* __restrict__ outV, float* __restrict__ outF,
                 bf16_t* __restrict__ outB, int M, int N, int K) {
  __shared__ bf16_t As[2][128 * 72];
  __shared__ bf16_t Bs[2][128 * 72];

  const int tid  = threadIdx.x;
  const int lane = tid & 31;
  const int wid  = tid >> 5;
  const int lhi  = lane >> 4;    // 0 | 1 (half-wave)
  const int llo  = lane & 15;
  const int wm   = (wid & 1) * 64;   // wave M offset in block tile
  const int wn   = (wid >> 1) * 32;  // wave N offset in block tile
  const int m0   = blockIdx.y * 128;
  const int n0   = blockIdx.x * 128;

  // cooperative loader: 128 rows x 8 16B-chunks per matrix; each thread does
  // rows arow+{0,32,64,96} at byte-chunk ach for both A and B = 8 async ops.
  const int arow = tid >> 3;         // 0..31
  const int ach  = (tid & 7) * 8;    // element offset 0..56

  v8f acc[4][2];
  #pragma unroll
  for (int mi = 0; mi < 4; mi++)
    #pragma unroll
    for (int ni = 0; ni < 2; ni++) acc[mi][ni] = (v8f){};

  auto issue_stage = [&](int k0, int buf) {
    #pragma unroll
    for (int i = 0; i < 4; i++) {
      const int row = arow + 32 * i;
      async_copy16(A  + (size_t)(m0 + row) * K + k0 + ach, &As[buf][row * 72 + ach]);
      async_copy16(Bt + (size_t)(n0 + row) * K + k0 + ach, &Bs[buf][row * 72 + ach]);
    }
  };

  issue_stage(0, 0);
  for (int k0 = 0; k0 < K; k0 += 64) {
    const int cur = (k0 >> 6) & 1;
    if (k0 + 64 < K) {
      issue_stage(k0 + 64, cur ^ 1);  // prefetch next stage into other buffer
      async_wait<8>();                // current stage's 8 copies complete
    } else {
      async_wait<0>();
    }
    __syncthreads();

    #pragma unroll
    for (int kk = 0; kk < 64; kk += 32) {
      // A fragments: lane holds row (llo), K halves split by lhi (ISA layout)
      Frag16 af[4];
      #pragma unroll
      for (int mi = 0; mi < 4; mi++) {
        const bf16_t* p = &As[cur][(wm + mi * 16 + llo) * 72 + kk];
        af[mi].u[0] = *(const u32x4*)(p + 8 * lhi);
        af[mi].u[1] = *(const u32x4*)(p + 16 + 8 * lhi);
      }
      // B fragments: lane holds column (llo), K halves split by lhi (ISA layout)
      Frag16 bfr[2];
      #pragma unroll
      for (int ni = 0; ni < 2; ni++) {
        const bf16_t* p = &Bs[cur][(wn + ni * 16 + llo) * 72 + kk];
        bfr[ni].u[0] = *(const u32x4*)(p + 8 * lhi);
        bfr[ni].u[1] = *(const u32x4*)(p + 16 + 8 * lhi);
      }
      #pragma unroll
      for (int mi = 0; mi < 4; mi++)
        #pragma unroll
        for (int ni = 0; ni < 2; ni++)
          acc[mi][ni] = wmma_bf16(af[mi], bfr[ni], acc[mi][ni]);
    }
    __syncthreads();
  }

  // Epilogue. C layout: VGPR r, lane L -> row = r + 8*(L>>4), col = L&15.
  #pragma unroll
  for (int mi = 0; mi < 4; mi++) {
    #pragma unroll
    for (int ni = 0; ni < 2; ni++) {
      const int gn  = n0 + wn + ni * 16 + llo;
      const float bia = bias[gn];
      #pragma unroll
      for (int r = 0; r < 8; r++) {
        const int gm = m0 + wm + mi * 16 + lhi * 8 + r;
        const float v = acc[mi][ni][r] + bia;
        if (EPI == 0) {
          const int part = gn >> 10, c = gn & 1023;
          const int h = c >> 6, dd = c & 63;
          const int b = gm >> 11, t = gm & (SEQ - 1);
          const size_t bh = (size_t)(b * N_HEADS + h);
          if (part == 0)      outQ[(bh * SEQ + t) * HEAD_DIM + dd] = (bf16_t)(v * 0.125f);
          else if (part == 1) outK[(bh * SEQ + t) * HEAD_DIM + dd] = (bf16_t)v;
          else                outV[(bh * HEAD_DIM + dd) * SEQ + t] = (bf16_t)v;
        } else if (EPI == 1) {
          const size_t idx = (size_t)gm * N + gn;
          outF[idx] = resid[idx] + v;
        } else {
          const float gl = 0.5f * v * (1.0f + erff(v * 0.70710678118654752f));
          outB[(size_t)gm * N + gn] = (bf16_t)gl;
        }
      }
    }
  }
}

// ---------------------------------------------------------------------------
// Flash attention: grid = (T/128, B*H). 8 waves/block, each wave owns a 16-row
// Q tile (full head_dim in registers, pre-scaled by 1/sqrt(hd) at QKV time).
// Streams 32-key K/V tiles through double-buffered async LDS staging, online
// softmax (wave32 ds_bpermute row reductions), QK^T and PV on WMMA bf16.
// V is pre-transposed ([B,H,64,T]) so PV B-fragments are two key-contiguous 16B loads/lane.
// ---------------------------------------------------------------------------
__global__ __launch_bounds__(256)
void attention_kernel(const bf16_t* __restrict__ Q, const bf16_t* __restrict__ Kc,
                      const bf16_t* __restrict__ Vt, bf16_t* __restrict__ Oc) {
  __shared__ bf16_t Ks[2][32 * 72];     // 32 keys x 64 dims (padded 72)
  __shared__ bf16_t Vs[2][64 * 40];     // 64 dims x 32 keys (padded 40)
  __shared__ bf16_t Ps[8][16 * 40];     // per-wave P tile relayout buffer

  const int tid  = threadIdx.x;
  const int lane = tid & 31, wid = tid >> 5;
  const int lhi  = lane >> 4, llo = lane & 15;
  const int bh    = blockIdx.y;                 // 0..31
  const int qbase = blockIdx.x * 128 + wid * 16;

  // Q fragments (A-layout): row = llo, hd split per half-wave
  const size_t qrow = ((size_t)bh * SEQ + qbase + llo) * HEAD_DIM;
  Frag16 qf[2];
  qf[0].u[0] = *(const u32x4*)(Q + qrow + 8 * lhi);
  qf[0].u[1] = *(const u32x4*)(Q + qrow + 16 + 8 * lhi);
  qf[1].u[0] = *(const u32x4*)(Q + qrow + 32 + 8 * lhi);
  qf[1].u[1] = *(const u32x4*)(Q + qrow + 48 + 8 * lhi);

  v8f accO[4];
  #pragma unroll
  for (int ni = 0; ni < 4; ni++) accO[ni] = (v8f){};
  float mrow[8], lrow[8];
  #pragma unroll
  for (int r = 0; r < 8; r++) { mrow[r] = -INFINITY; lrow[r] = 0.0f; }

  const int kkey = tid >> 3, koff = (tid & 7) * 8;   // K tile: 32 x 8 chunks
  const int vrow = tid >> 2, voff = (tid & 3) * 8;   // V tile: 64 x 4 chunks

  auto issue_kv = [&](int j, int buf) {
    async_copy16(Kc + ((size_t)bh * SEQ + j + kkey) * HEAD_DIM + koff,
                 &Ks[buf][kkey * 72 + koff]);
    async_copy16(Vt + ((size_t)bh * HEAD_DIM + vrow) * SEQ + j + voff,
                 &Vs[buf][vrow * 40 + voff]);
  };

  issue_kv(0, 0);
  for (int j = 0; j < SEQ; j += 32) {
    const int cur = (j >> 5) & 1;
    if (j + 32 < SEQ) {
      issue_kv(j + 32, cur ^ 1);
      async_wait<2>();
    } else {
      async_wait<0>();
    }
    __syncthreads();

    // S = Q K^T (two 16-key column tiles, K-dim = head_dim in 2 WMMAs)
    v8f s[2];
    #pragma unroll
    for (int ni = 0; ni < 2; ni++) {
      Frag16 kb0, kb1;
      const bf16_t* kp = &Ks[cur][(ni * 16 + llo) * 72 + 8 * lhi];
      kb0.u[0] = *(const u32x4*)(kp);      kb0.u[1] = *(const u32x4*)(kp + 16);
      kb1.u[0] = *(const u32x4*)(kp + 32); kb1.u[1] = *(const u32x4*)(kp + 48);
      v8f z = (v8f){};
      z = wmma_bf16(qf[0], kb0, z);
      z = wmma_bf16(qf[1], kb1, z);
      s[ni] = z;
    }

    // online softmax, row stats live per-lane (replicated across half-wave)
    #pragma unroll
    for (int r = 0; r < 8; r++) {
      const float s0 = s[0][r];
      const float s1 = s[1][r];
      float tmax = fmaxf(s0, s1);
      #pragma unroll
      for (int o = 1; o < 16; o <<= 1) tmax = fmaxf(tmax, __shfl_xor(tmax, o, 32));
      const float mnew = fmaxf(mrow[r], tmax);
      const float fr = __expf(mrow[r] - mnew);
      const float p0 = __expf(s0 - mnew);
      const float p1 = __expf(s1 - mnew);
      float ps = p0 + p1;
      #pragma unroll
      for (int o = 1; o < 16; o <<= 1) ps += __shfl_xor(ps, o, 32);
      lrow[r] = lrow[r] * fr + ps;
      mrow[r] = mnew;
      #pragma unroll
      for (int ni = 0; ni < 4; ni++) accO[ni][r] *= fr;
      bf16_t* pp = &Ps[wid][(lhi * 8 + r) * 40];
      pp[llo]      = (bf16_t)p0;
      pp[16 + llo] = (bf16_t)p1;
    }

    // relayout P (C-layout) -> A-fragment via wave-private LDS bounce
    Frag16 pf;
    const bf16_t* pr = &Ps[wid][llo * 40];
    pf.u[0] = *(const u32x4*)(pr + 8 * lhi);
    pf.u[1] = *(const u32x4*)(pr + 16 + 8 * lhi);

    // O += P V  (4 head-dim column tiles)
    #pragma unroll
    for (int ni = 0; ni < 4; ni++) {
      Frag16 vb;
      const bf16_t* vp = &Vs[cur][(ni * 16 + llo) * 40 + 8 * lhi];
      vb.u[0] = *(const u32x4*)(vp);
      vb.u[1] = *(const u32x4*)(vp + 16);
      accO[ni] = wmma_bf16(pf, vb, accO[ni]);
    }
    __syncthreads();
  }

  // normalize + write context rows: token row = b*SEQ + q, col = h*64 + dim
  const int b = bh >> 4, h = bh & 15;
  #pragma unroll
  for (int r = 0; r < 8; r++) {
    const float inv = 1.0f / lrow[r];
    const int qr = lhi * 8 + r;
    const size_t rowo = ((size_t)b * SEQ + qbase + qr) * D_MODEL + h * HEAD_DIM;
    #pragma unroll
    for (int ni = 0; ni < 4; ni++)
      Oc[rowo + ni * 16 + llo] = (bf16_t)(accO[ni][r] * inv);
  }
}

// ---------------------------------------------------------------------------
extern "C" void kernel_launch(void* const* d_in, const int* in_sizes, int n_in,
                              void* d_out, int out_size, void* d_ws, size_t ws_size,
                              hipStream_t stream) {
  (void)in_sizes; (void)n_in; (void)out_size; (void)ws_size;
  const float* x     = (const float*)d_in[0];
  const float* n1g   = (const float*)d_in[1];
  const float* n1b   = (const float*)d_in[2];
  const float* qkv_w = (const float*)d_in[3];
  const float* qkv_b = (const float*)d_in[4];
  const float* out_w = (const float*)d_in[5];
  const float* out_b = (const float*)d_in[6];
  const float* n2g   = (const float*)d_in[7];
  const float* n2b   = (const float*)d_in[8];
  const float* ff1_w = (const float*)d_in[9];
  const float* ff1_b = (const float*)d_in[10];
  const float* ff2_w = (const float*)d_in[11];
  const float* ff2_b = (const float*)d_in[12];

  char* ws = (char*)d_ws;
  size_t off = 0;
  auto alloc = [&](size_t bytes) -> void* {
    void* p = ws + off;
    off += (bytes + 255) & ~(size_t)255;
    return p;
  };

  bf16_t* wqkvT = (bf16_t*)alloc((size_t)3 * D_MODEL * D_MODEL * 2);
  bf16_t* woutT = (bf16_t*)alloc((size_t)D_MODEL * D_MODEL * 2);
  bf16_t* wff1T = (bf16_t*)alloc((size_t)FF_DIM * D_MODEL * 2);
  bf16_t* wff2T = (bf16_t*)alloc((size_t)D_MODEL * FF_DIM * 2);
  bf16_t* h1    = (bf16_t*)alloc((size_t)MTOK * D_MODEL * 2);
  bf16_t* Qb    = (bf16_t*)alloc((size_t)MTOK * D_MODEL * 2);
  bf16_t* Kb    = (bf16_t*)alloc((size_t)MTOK * D_MODEL * 2);
  bf16_t* Vtb   = (bf16_t*)alloc((size_t)MTOK * D_MODEL * 2);
  bf16_t* attn  = (bf16_t*)alloc((size_t)MTOK * D_MODEL * 2);
  float*  xres  = (float*) alloc((size_t)MTOK * D_MODEL * 4);
  bf16_t* h2    = (bf16_t*)alloc((size_t)MTOK * D_MODEL * 2);
  bf16_t* hff   = (bf16_t*)alloc((size_t)MTOK * FF_DIM * 2);

  const dim3 tb(32, 8);
  // weight transposes (f32 -> bf16, K-contiguous)
  transpose_cast_kernel<<<dim3(3 * D_MODEL / 32, D_MODEL / 32), tb, 0, stream>>>(qkv_w, wqkvT, D_MODEL, 3 * D_MODEL);
  transpose_cast_kernel<<<dim3(D_MODEL / 32, D_MODEL / 32),     tb, 0, stream>>>(out_w, woutT, D_MODEL, D_MODEL);
  transpose_cast_kernel<<<dim3(FF_DIM / 32, D_MODEL / 32),      tb, 0, stream>>>(ff1_w, wff1T, D_MODEL, FF_DIM);
  transpose_cast_kernel<<<dim3(D_MODEL / 32, FF_DIM / 32),      tb, 0, stream>>>(ff2_w, wff2T, FF_DIM, D_MODEL);

  // LN1 -> h1
  ln_cast_kernel<<<MTOK, 256, 0, stream>>>(x, n1g, n1b, h1);

  // QKV GEMM with head scatter (V transposed, Q pre-scaled)
  gemm_kernel<0><<<dim3(3 * D_MODEL / 128, MTOK / 128), 256, 0, stream>>>(
      h1, wqkvT, qkv_b, nullptr, Qb, Kb, Vtb, nullptr, nullptr,
      MTOK, 3 * D_MODEL, D_MODEL);

  // flash attention
  attention_kernel<<<dim3(SEQ / 128, BATCH * N_HEADS), 256, 0, stream>>>(Qb, Kb, Vtb, attn);

  // out-proj + residual (f32)
  gemm_kernel<1><<<dim3(D_MODEL / 128, MTOK / 128), 256, 0, stream>>>(
      attn, woutT, out_b, x, nullptr, nullptr, nullptr, xres, nullptr,
      MTOK, D_MODEL, D_MODEL);

  // LN2 -> h2
  ln_cast_kernel<<<MTOK, 256, 0, stream>>>(xres, n2g, n2b, h2);

  // FF1 + exact GELU (bf16)
  gemm_kernel<2><<<dim3(FF_DIM / 128, MTOK / 128), 256, 0, stream>>>(
      h2, wff1T, ff1_b, nullptr, nullptr, nullptr, nullptr, nullptr, hff,
      MTOK, FF_DIM, D_MODEL);

  // FF2 + residual -> d_out (f32)
  gemm_kernel<1><<<dim3(D_MODEL / 128, MTOK / 128), 256, 0, stream>>>(
      hff, wff2T, ff2_b, xres, nullptr, nullptr, nullptr, (float*)d_out, nullptr,
      MTOK, D_MODEL, FF_DIM);
}
